// TransformerDecoder_7086696038688
// MI455X (gfx1250) — matrix operand order repaired and checked by reading, not compiled, not run on hardware
//
#include <hip/hip_runtime.h>

// ---------------------------------------------------------------------------
// TransformerDecoder forward for MI455X (gfx1250), wave32 + WMMA f16->f32.
// One strided-batched WMMA GEMM does all heavy math. Double-buffered LDS,
// float4 global staging with fp32->f16 conversion, B stored transposed in LDS
// so both A and B fragments are contiguous 32B ds_load_b128 pairs.
// Dims: N=16 T=512 V=10000 DIN=2048 D=1024 H=16 L=4 FF=2048 DH=64
// NOTE: all GEMM calls here have K % 32 == 0 (no K guards needed).
// ---------------------------------------------------------------------------

typedef __attribute__((ext_vector_type(16))) _Float16 v16h;
typedef __attribute__((ext_vector_type(8)))  _Float16 h8;
typedef __attribute__((ext_vector_type(8)))  float    v8f;

#define BM 128
#define BN 128
#define BK 32
#define AKP (BK + 8)   // As row pitch in halves (80B, 16B-aligned slices)
#define BKP (BK + 8)   // Bs row pitch in halves (B kept transposed: [n][k])

// ---------------------------------------------------------------------------
// C[b] = alpha * A[b] x B[b] (+ bias) (opt ReLU)
// A: MxK row-major (lda). B: KxN row-major (ldb), or NxK row-major if transB.
// Batch z -> (bn = z/batchH, bh = z%batchH) with independent strides.
// Block tile 128x128x32; 8 waves; each wave 4x2 WMMA 16x16x32 tiles.
// ---------------------------------------------------------------------------
__global__ __launch_bounds__(256)
void gemm_wmma_f16(const float* __restrict__ A, const float* __restrict__ B,
                   float* __restrict__ C, const float* __restrict__ bias,
                   int M, int N, int K, int lda, int ldb, int ldc,
                   long sAn, long sAh, long sBn, long sBh, long sCn, long sCh,
                   int batchH, int transB, float alpha, int relu)
{
    __shared__ _Float16 As[2][BM][AKP];
    __shared__ _Float16 Bs[2][BN][BKP];

    const int z  = blockIdx.z;
    const int bn = z / batchH, bh = z % batchH;
    A += (long)bn * sAn + (long)bh * sAh;
    B += (long)bn * sBn + (long)bh * sBh;
    C += (long)bn * sCn + (long)bh * sCh;

    const int blockM = blockIdx.y * BM;
    const int blockN = blockIdx.x * BN;
    const int t    = threadIdx.x;
    const int lane = t & 31;
    const int wid  = t >> 5;
    const int waveM = (wid >> 2) * 64;  // 0 or 64
    const int waveN = (wid & 3) * 32;   // 0..96
    const int l15   = lane & 15;
    const int k0a   = (lane < 16) ? 0 : 16;  // A frag K-group (contiguous, matches B)
    const int k0b   = (lane < 16) ? 0 : 16;  // B frag K-group

    const bool fullM = (blockM + BM <= M);
    const bool fullN = (blockN + BN <= N);

    // staging thread mappings
    const int ar  = t >> 1, acb = (t & 1) * 16;        // A: row, col-base (16 floats)
    const int bk0 = t >> 3, bnb = (t & 7) * 16;        // B !trans: k, n-base
    // B trans: n = t>>1, k-base = (t&1)*16 (same shape as A mapping)

    float arg[16], brg[16];

    // ---- global -> registers ----
    auto loadA = [&](int kk) {
        const int grow = blockM + ar;
        if (fullM) {
            const float4* p = (const float4*)(A + (long)grow * lda + kk + acb);
#pragma unroll
            for (int i = 0; i < 4; ++i) {
                float4 f = p[i];
                arg[4*i+0] = f.x; arg[4*i+1] = f.y; arg[4*i+2] = f.z; arg[4*i+3] = f.w;
            }
        } else {
            const int cr = grow < M ? grow : (M - 1);
            const float* p = A + (long)cr * lda + kk + acb;
#pragma unroll
            for (int i = 0; i < 16; ++i) {
                float v = p[i];                     // address always valid (clamped row)
                arg[i] = (grow < M) ? v : 0.0f;
            }
        }
    };
    auto loadB = [&](int kk) {
        if (transB) {
            const int gn = blockN + ar;             // reuse A-shaped mapping
            if (fullN) {
                const float4* p = (const float4*)(B + (long)gn * ldb + kk + acb);
#pragma unroll
                for (int i = 0; i < 4; ++i) {
                    float4 f = p[i];
                    brg[4*i+0] = f.x; brg[4*i+1] = f.y; brg[4*i+2] = f.z; brg[4*i+3] = f.w;
                }
            } else {
                const int cn = gn < N ? gn : (N - 1);
                const float* p = B + (long)cn * ldb + kk + acb;
#pragma unroll
                for (int i = 0; i < 16; ++i) {
                    float v = p[i];
                    brg[i] = (gn < N) ? v : 0.0f;
                }
            }
        } else {
            const int gk = kk + bk0;                // in range: K % BK == 0
            if (fullN) {
                const float4* p = (const float4*)(B + (long)gk * ldb + blockN + bnb);
#pragma unroll
                for (int i = 0; i < 4; ++i) {
                    float4 f = p[i];
                    brg[4*i+0] = f.x; brg[4*i+1] = f.y; brg[4*i+2] = f.z; brg[4*i+3] = f.w;
                }
            } else {
                const float* p = B + (long)gk * ldb;
#pragma unroll
                for (int i = 0; i < 16; ++i) {
                    const int gn = blockN + bnb + i;
                    const int cn = gn < N ? gn : (N - 1);
                    float v = p[cn];
                    brg[i] = (gn < N) ? v : 0.0f;
                }
            }
        }
    };
    // ---- registers -> LDS (with fp32->f16 convert) ----
    auto stageA = [&](int buf) {
        h8 h0, h1;
#pragma unroll
        for (int j = 0; j < 8; ++j) { h0[j] = (_Float16)arg[j]; h1[j] = (_Float16)arg[8+j]; }
        *(h8*)&As[buf][ar][acb]     = h0;
        *(h8*)&As[buf][ar][acb + 8] = h1;
    };
    auto stageB = [&](int buf) {
        if (transB) {                               // contiguous: Bs[n][kbase..]
            h8 h0, h1;
#pragma unroll
            for (int j = 0; j < 8; ++j) { h0[j] = (_Float16)brg[j]; h1[j] = (_Float16)brg[8+j]; }
            *(h8*)&Bs[buf][ar][acb]     = h0;
            *(h8*)&Bs[buf][ar][acb + 8] = h1;
        } else {                                    // scatter: Bs[nbase+i][k]
#pragma unroll
            for (int i = 0; i < 16; ++i)
                Bs[buf][bnb + i][bk0] = (_Float16)brg[i];
        }
    };

    v8f acc[4][2] = {};

    loadA(0); loadB(0);
    stageA(0); stageB(0);
    int cur = 0;

    for (int kk = 0; kk < K; kk += BK) {
        __syncthreads();
        const bool hasNext = (kk + BK) < K;
        if (hasNext) { loadA(kk + BK); loadB(kk + BK); }   // overlap with WMMAs below

        // B fragments: 32 contiguous bytes per lane
        v16h bf[2];
#pragma unroll
        for (int nt = 0; nt < 2; ++nt)
            bf[nt] = *(const v16h*)&Bs[cur][waveN + nt * 16 + l15][k0b];

#pragma unroll
        for (int mt = 0; mt < 4; ++mt) {
            const _Float16* arow = &As[cur][waveM + mt * 16 + l15][0];
            h8 a0 = *(const h8*)(arow + k0a);
            h8 a1 = *(const h8*)(arow + k0a + 8);
            v16h af;
#pragma unroll
            for (int j = 0; j < 8; ++j) { af[j] = a0[j]; af[j + 8] = a1[j]; }
#pragma unroll
            for (int nt = 0; nt < 2; ++nt)
                acc[mt][nt] = __builtin_amdgcn_wmma_f32_16x16x32_f16(
                    false, af, false, bf[nt], (short)0, acc[mt][nt], false, false);
        }

        if (hasNext) {
            __syncthreads();
            stageA(cur ^ 1); stageB(cur ^ 1);
            cur ^= 1;
        }
    }

    // ---- epilogue (C/D layout: lanes 0-15 -> M=r, lanes 16-31 -> M=r+8) ----
    const int rowOff = (lane < 16) ? 0 : 8;
    if (fullM && fullN) {
#pragma unroll
        for (int nt = 0; nt < 2; ++nt) {
            const int colG = blockN + waveN + nt * 16 + l15;
            const float bv = bias ? bias[colG] : 0.0f;
#pragma unroll
            for (int mt = 0; mt < 4; ++mt) {
#pragma unroll
                for (int r = 0; r < 8; ++r) {
                    const int rowG = blockM + waveM + mt * 16 + r + rowOff;
                    float v = alpha * acc[mt][nt][r] + bv;
                    if (relu) v = v > 0.0f ? v : 0.0f;
                    C[(long)rowG * ldc + colG] = v;
                }
            }
        }
    } else {
#pragma unroll
        for (int nt = 0; nt < 2; ++nt) {
            const int colG = blockN + waveN + nt * 16 + l15;
            if (colG >= N) continue;
            const float bv = bias ? bias[colG] : 0.0f;
#pragma unroll
            for (int mt = 0; mt < 4; ++mt) {
#pragma unroll
                for (int r = 0; r < 8; ++r) {
                    const int rowG = blockM + waveM + mt * 16 + r + rowOff;
                    if (rowG < M) {
                        float v = alpha * acc[mt][nt][r] + bv;
                        if (relu) v = v > 0.0f ? v : 0.0f;
                        C[(long)rowG * ldc + colG] = v;
                    }
                }
            }
        }
    }
}

// ---------------------------------------------------------------------------
// out[n*T+t, :] = cap_emb[captions[n,t]] + pos_emb[t]
// ---------------------------------------------------------------------------
__global__ __launch_bounds__(256)
void embed_kernel(const int* __restrict__ caps, const float* __restrict__ cap_emb,
                  const float* __restrict__ pos_emb, float* __restrict__ out,
                  int Tdim, int Ddim)
{
    const long row = blockIdx.x;          // n*T + t
    const int  tt  = (int)(row % Tdim);
    const int  tok = caps[row];
    const long base = row * Ddim;
    for (int d = threadIdx.x; d < Ddim; d += 256)
        out[base + d] = cap_emb[(long)tok * Ddim + d] + pos_emb[(long)tt * Ddim + d];
}

// ---------------------------------------------------------------------------
// In-place causal softmax per row. blockIdx.x = s (row in T), blockIdx.y = (n*H+h)
// ---------------------------------------------------------------------------
__global__ __launch_bounds__(128)
void softmax_causal(float* __restrict__ S, int Tdim)
{
    const int  s    = blockIdx.x;
    const long base = ((long)blockIdx.y * Tdim + s) * (long)Tdim;
    __shared__ float red[128];
    const int t = threadIdx.x;

    float mx = -1e30f;
    for (int c = t; c < Tdim; c += 128) {
        float v = S[base + c];
        if (c > s) v = -1e8f;
        S[base + c] = v;
        mx = fmaxf(mx, v);
    }
    red[t] = mx; __syncthreads();
    for (int o = 64; o > 0; o >>= 1) { if (t < o) red[t] = fmaxf(red[t], red[t + o]); __syncthreads(); }
    mx = red[0]; __syncthreads();

    float sum = 0.0f;
    for (int c = t; c < Tdim; c += 128) {
        float e = __expf(S[base + c] - mx);
        S[base + c] = e; sum += e;
    }
    red[t] = sum; __syncthreads();
    for (int o = 64; o > 0; o >>= 1) { if (t < o) red[t] += red[t + o]; __syncthreads(); }
    const float inv = 1.0f / red[0];
    for (int c = t; c < Tdim; c += 128) S[base + c] *= inv;
}

// ---------------------------------------------------------------------------
// out = LayerNorm(out + x) * g + b   (in place on `out`), one block per row
// ---------------------------------------------------------------------------
__global__ __launch_bounds__(256)
void add_ln_kernel(float* __restrict__ out, const float* __restrict__ x,
                   const float* __restrict__ g, const float* __restrict__ b, int Ddim)
{
    const long base = (long)blockIdx.x * Ddim;
    const int  t    = threadIdx.x;
    __shared__ float rs[256], rq[256];
    float s = 0.0f, q = 0.0f;
    for (int d = t; d < Ddim; d += 256) {
        const float v = out[base + d] + x[base + d];
        s += v; q += v * v;
    }
    rs[t] = s; rq[t] = q; __syncthreads();
    for (int o = 128; o > 0; o >>= 1) {
        if (t < o) { rs[t] += rs[t + o]; rq[t] += rq[t + o]; }
        __syncthreads();
    }
    const float mean = rs[0] / Ddim;
    const float var  = rq[0] / Ddim - mean * mean;
    const float inv  = rsqrtf(var + 1e-5f);
    for (int d = t; d < Ddim; d += 256) {
        const float v = out[base + d] + x[base + d];
        out[base + d] = g[d] * (v - mean) * inv + b[d];
    }
}

// ---------------------------------------------------------------------------
// Cross-attn with a single K/V token: softmax == 1, so pre-Wo output is a
// broadcast of vh over T, in (N,H,T,Dh)-contiguous order (reference's reshape
// quirk is free when this buffer is read back as (N*T, D)).
// ---------------------------------------------------------------------------
__global__ __launch_bounds__(256)
void bcast_cross(const float* __restrict__ vh, float* __restrict__ y,
                 int Hdim, int Tdim, int DHdim, int Ddim, long total)
{
    const long i = (long)blockIdx.x * 256 + threadIdx.x;
    if (i >= total) return;
    const int  dh = (int)(i % DHdim);
    const long r  = i / DHdim;
    const long r2 = r / Tdim;            // t discarded: broadcast over T
    const int  h  = (int)(r2 % Hdim);
    const long n  = r2 / Hdim;
    y[i] = vh[n * Ddim + (long)h * DHdim + dh];
}

// ---------------------------------------------------------------------------
// Host side
// ---------------------------------------------------------------------------
static inline void launch_gemm(hipStream_t st, const float* A, const float* B, float* C,
                               const float* bias, int M, int N, int K,
                               int lda, int ldb, int ldc,
                               long sAn, long sAh, long sBn, long sBh, long sCn, long sCh,
                               int batchN, int batchH, int transB, float alpha, int relu)
{
    dim3 grid((N + BN - 1) / BN, (M + BM - 1) / BM, batchN * batchH);
    gemm_wmma_f16<<<grid, 256, 0, st>>>(A, B, C, bias, M, N, K, lda, ldb, ldc,
                                        sAn, sAh, sBn, sBh, sCn, sCh,
                                        batchH, transB, alpha, relu);
}

extern "C" void kernel_launch(void* const* d_in, const int* in_sizes, int n_in,
                              void* d_out, int out_size, void* d_ws, size_t ws_size,
                              hipStream_t stream)
{
    (void)in_sizes; (void)n_in; (void)out_size; (void)ws_size;
    enum { Nb = 16, T = 512, V = 10000, DIN = 2048, D = 1024, H = 16, L = 4, FF = 2048, DH = 64 };
    const long NT = (long)Nb * T;  // 8192 rows

    const float* features = (const float*)d_in[0];
    const int*   captions = (const int*)  d_in[1];
    const float* feat_W   = (const float*)d_in[2];
    const float* feat_b   = (const float*)d_in[3];
    const float* cap_emb  = (const float*)d_in[4];
    const float* pos_emb  = (const float*)d_in[5];
    const float* sa_Wq = (const float*)d_in[6],  *sa_Wk = (const float*)d_in[7];
    const float* sa_Wv = (const float*)d_in[8],  *sa_Wo = (const float*)d_in[9];
    const float* sa_bo = (const float*)d_in[10], *sa_g  = (const float*)d_in[11];
    const float* sa_b  = (const float*)d_in[12];
    const float* ca_Wv = (const float*)d_in[15], *ca_Wo = (const float*)d_in[16];
    const float* ca_bo = (const float*)d_in[17], *ca_g  = (const float*)d_in[18];
    const float* ca_b  = (const float*)d_in[19];
    const float* ff_W1 = (const float*)d_in[20], *ff_b1 = (const float*)d_in[21];
    const float* ff_W2 = (const float*)d_in[22], *ff_b2 = (const float*)d_in[23];
    const float* ff_g  = (const float*)d_in[24], *ff_b  = (const float*)d_in[25];
    const float* score_W = (const float*)d_in[26], *score_b = (const float*)d_in[27];

    // workspace carve-up (floats)
    float* ws  = (float*)d_ws;
    const long SZ = NT * D;                 // 8388608
    float* OUT = ws;                        // running hidden state (N*T, D)
    float* XB  = ws + 1 * SZ;               // sublayer output
    float* QB  = ws + 2 * SZ;
    float* KB  = ws + 3 * SZ;
    float* VB  = ws + 4 * SZ;
    float* YB  = ws + 5 * SZ;               // attn output, (N,H,T,Dh) contiguous
    float* HB  = ws + 6 * SZ;               // FF hidden (N*T, FF) = 2*SZ
    float* SB  = ws + 8 * SZ;               // scores (N,H,T,T)
    float* FE  = SB + (long)Nb * H * T * T; // (16, D)
    float* VH  = FE + (long)Nb * D;         // (16, D)

    // fe = features @ feat_W + feat_b
    launch_gemm(stream, features, feat_W, FE, feat_b, Nb, D, DIN, DIN, D, D,
                0, 0, 0, 0, 0, 0, 1, 1, 0, 1.0f, 0);

    // out = cap_emb[captions] + pos_emb
    embed_kernel<<<dim3((unsigned)NT), 256, 0, stream>>>(captions, cap_emb, pos_emb, OUT, T, D);

    const float inv_sqrt_dh = 0.125f;  // 1/sqrt(64)

    for (int i = 0; i < L; ++i) {
        const long wOff = (long)i * D * D, bOff = (long)i * D;

        // ---- self attention ----
        launch_gemm(stream, OUT, sa_Wq + wOff, QB, nullptr, (int)NT, D, D, D, D, D,
                    0, 0, 0, 0, 0, 0, 1, 1, 0, 1.0f, 0);
        launch_gemm(stream, OUT, sa_Wk + wOff, KB, nullptr, (int)NT, D, D, D, D, D,
                    0, 0, 0, 0, 0, 0, 1, 1, 0, 1.0f, 0);
        launch_gemm(stream, OUT, sa_Wv + wOff, VB, nullptr, (int)NT, D, D, D, D, D,
                    0, 0, 0, 0, 0, 0, 1, 1, 0, 1.0f, 0);

        // scores[n,h] = (Q_nh @ K_nh^T) / sqrt(DH);  batched over (n,h)
        launch_gemm(stream, QB, KB, SB, nullptr, T, T, DH, D, D, T,
                    (long)T * D, DH, (long)T * D, DH, (long)H * T * T, (long)T * T,
                    Nb, H, /*transB=*/1, inv_sqrt_dh, 0);

        softmax_causal<<<dim3(T, Nb * H), 128, 0, stream>>>(SB, T);

        // y[n,h] = attn @ V_nh  -> (N,H,T,Dh) contiguous (reshape quirk for free)
        launch_gemm(stream, SB, VB, YB, nullptr, T, DH, T, T, D, DH,
                    (long)H * T * T, (long)T * T, (long)T * D, DH,
                    (long)H * T * DH, (long)T * DH, Nb, H, 0, 1.0f, 0);

        launch_gemm(stream, YB, sa_Wo + wOff, XB, sa_bo + bOff, (int)NT, D, D, D, D, D,
                    0, 0, 0, 0, 0, 0, 1, 1, 0, 1.0f, 0);
        add_ln_kernel<<<dim3((unsigned)NT), 256, 0, stream>>>(OUT, XB, sa_g + bOff, sa_b + bOff, D);

        // ---- cross attention (single K/V token -> softmax==1 -> broadcast of vh) ----
        launch_gemm(stream, FE, ca_Wv + wOff, VH, nullptr, Nb, D, D, D, D, D,
                    0, 0, 0, 0, 0, 0, 1, 1, 0, 1.0f, 0);
        {
            const long total = (long)Nb * H * T * DH;
            bcast_cross<<<dim3((unsigned)((total + 255) / 256)), 256, 0, stream>>>(
                VH, YB, H, T, DH, D, total);
        }
        launch_gemm(stream, YB, ca_Wo + wOff, XB, ca_bo + bOff, (int)NT, D, D, D, D, D,
                    0, 0, 0, 0, 0, 0, 1, 1, 0, 1.0f, 0);
        add_ln_kernel<<<dim3((unsigned)NT), 256, 0, stream>>>(OUT, XB, ca_g + bOff, ca_b + bOff, D);

        // ---- feed forward ----
        launch_gemm(stream, OUT, ff_W1 + (long)i * D * FF, HB, ff_b1 + (long)i * FF,
                    (int)NT, FF, D, D, FF, FF, 0, 0, 0, 0, 0, 0, 1, 1, 0, 1.0f, /*relu=*/1);
        launch_gemm(stream, HB, ff_W2 + (long)i * FF * D, XB, ff_b2 + bOff,
                    (int)NT, D, FF, FF, D, D, 0, 0, 0, 0, 0, 0, 1, 1, 0, 1.0f, 0);
        add_ln_kernel<<<dim3((unsigned)NT), 256, 0, stream>>>(OUT, XB, ff_g + bOff, ff_b + bOff, D);
    }

    // final vocab projection: (N*T, D) @ (D, V) + score_b -> d_out
    launch_gemm(stream, OUT, score_W, (float*)d_out, score_b, (int)NT, V, D, D, V, V,
                0, 0, 0, 0, 0, 0, 1, 1, 0, 1.0f, 0);
}
